// SparseGCN_62122406969952
// MI455X (gfx1250) — matrix operand order repaired and checked by reading, not compiled, not run on hardware
//
#include <hip/hip_runtime.h>
#include <stdint.h>

// CDNA5 / gfx1250: wave32, WMMA 16x16x32 bf16 -> f32 accumulate.
typedef __attribute__((ext_vector_type(16))) __bf16 bf16x16;
typedef __attribute__((ext_vector_type(8)))  float  f32x8;

#define F    128
#define DEG  16
#define OUTD 128
#define K2   256            // 2*F (concat width)
#define KPAD 264            // padded LDS row: 132 dwords -> min-conflict b128 reads

struct Q2 { uint4 a, b; };  // 32B container to bit-cast into bf16x16

__device__ __forceinline__ unsigned short f2bf(float f) {
    unsigned u = __builtin_bit_cast(unsigned, f);
    u += 0x7FFFu + ((u >> 16) & 1u);            // round-to-nearest-even
    return (unsigned short)(u >> 16);
}
__device__ __forceinline__ unsigned pack2(float lo, float hi) {
    return (unsigned)f2bf(lo) | ((unsigned)f2bf(hi) << 16);
}

// ---- one-time prep: W[k][n] fp32 -> Wt[n][k] bf16 (64 KB, L2-resident) ----
__global__ __launch_bounds__(256) void wt_prep_kernel(const float* __restrict__ w,
                                                      unsigned short* __restrict__ wtg) {
    const int i = blockIdx.x * 256 + threadIdx.x;   // 0..K2*OUTD-1 (coalesced read)
    const int k = i >> 7;
    const int n = i & 127;
    wtg[n * K2 + k] = f2bf(w[i]);
}

// ---- fused gather-average + [x|h] @ W + bias + sigmoid ----
template <bool USE_WS>
__global__ __launch_bounds__(256) void gcn_fused_kernel(
    const float*          __restrict__ x,    // [N,F]
    const int*            __restrict__ adj,  // [E,2] rows = (dst, src)
    const unsigned short* __restrict__ wtg,  // bf16 Wt[OUTD][K2] in d_ws (USE_WS)
    const float*          __restrict__ w,    // fp32 W[K2][OUTD] (fallback path)
    const float*          __restrict__ bias, // [OUTD]
    float*                __restrict__ out,  // [N,OUTD]
    int n_nodes)
{
    __shared__ __attribute__((aligned(16))) unsigned short at[16][KPAD];   // ~8.3 KB
    __shared__ __attribute__((aligned(16)))
        unsigned short wl[USE_WS ? 1 : OUTD][USE_WS ? 1 : KPAD];           // fallback only

    const int t         = threadIdx.x;          // 0..255
    const int lane      = t & 31;
    const int wave      = t >> 5;               // 0..7
    const int node_base = blockIdx.x * 16;

    const int m    = t >> 4;                    // node within tile (0..15)
    const int part = t & 15;                    // feature-chunk / edge slot
    const int fb   = part * 8;
    const int node = node_base + m;

    // One adjacency index per thread (coalesced); broadcast later via wave shuffle.
    const size_t e      = (size_t)node + (size_t)part * (size_t)n_nodes;
    const int    my_src = adj[2 * e + 1];

    if constexpr (USE_WS) {
        // Warm L2 with this wave's weight rows while we gather.
        __builtin_prefetch(wtg + (size_t)(wave * 16 + part) * K2, 0, 1);
    } else {
        // Fallback: stage W into LDS (transpose + cvt bf16).
        for (int i = t; i < K2 * OUTD; i += 256) {
            const int k = i >> 7;
            const int n = i & 127;
            wl[n][k] = f2bf(w[i]);
        }
    }

    // ---- gather + average 16 neighbors; build A = [x | h_neigh] in LDS ----
    {
        const float4 s0 = *(const float4*)(x + (size_t)node * F + fb);
        const float4 s1 = *(const float4*)(x + (size_t)node * F + fb + 4);

        float acc[8];
        #pragma unroll
        for (int i = 0; i < 8; ++i) acc[i] = 0.0f;

        #pragma unroll
        for (int k = 0; k < DEG; ++k) {
            // lanes 0-15 hold node 2w's edges, lanes 16-31 hold node 2w+1's
            const int s = __shfl(my_src, (lane & 16) | k, 32);
            const float4* p = (const float4*)(x + (size_t)s * F + fb);
            const float4 b0 = p[0];
            const float4 b1 = p[1];
            acc[0] += b0.x; acc[1] += b0.y; acc[2] += b0.z; acc[3] += b0.w;
            acc[4] += b1.x; acc[5] += b1.y; acc[6] += b1.z; acc[7] += b1.w;
        }
        const float inv = 1.0f / (float)DEG;

        unsigned* arow = (unsigned*)&at[m][fb];        // x part
        arow[0] = pack2(s0.x, s0.y);
        arow[1] = pack2(s0.z, s0.w);
        arow[2] = pack2(s1.x, s1.y);
        arow[3] = pack2(s1.z, s1.w);
        unsigned* hrow = (unsigned*)&at[m][F + fb];    // h_neigh part
        hrow[0] = pack2(acc[0] * inv, acc[1] * inv);
        hrow[1] = pack2(acc[2] * inv, acc[3] * inv);
        hrow[2] = pack2(acc[4] * inv, acc[5] * inv);
        hrow[3] = pack2(acc[6] * inv, acc[7] * inv);
    }
    __syncthreads();

    // ---- WMMA: wave w owns output cols 16w..16w+15, K = 256 in 8 steps ----
    const int nl = lane & 15;
    const int hi = lane >> 4;
    const int n0 = wave * 16;
    const unsigned short* brow =
        USE_WS ? (wtg + (size_t)(n0 + nl) * K2) : (const unsigned short*)nullptr;

    f32x8 c = {0.f, 0.f, 0.f, 0.f, 0.f, 0.f, 0.f, 0.f};

    #pragma unroll
    for (int kk = 0; kk < K2; kk += 32) {
        // A 16x32 bf16 fragment: lane M = nl; K = akb..akb+15 contiguous
        const int akb = kk + hi * 16;
        Q2 qa;
        qa.a = *(const uint4*)&at[nl][akb];
        qa.b = *(const uint4*)&at[nl][akb + 8];
        const bf16x16 av = __builtin_bit_cast(bf16x16, qa);

        // B 32x16 bf16 fragment: lane N = nl; K = bkb..bkb+15 contiguous in Wt[n][k]
        const int bkb = kk + hi * 16;
        Q2 qb;
        if constexpr (USE_WS) {
            qb.a = *(const uint4*)(brow + bkb);          // L2-resident global
            qb.b = *(const uint4*)(brow + bkb + 8);
        } else {
            qb.a = *(const uint4*)&wl[n0 + nl][bkb];
            qb.b = *(const uint4*)&wl[n0 + nl][bkb + 8];
        }
        const bf16x16 bv = __builtin_bit_cast(bf16x16, qb);

        c = __builtin_amdgcn_wmma_f32_16x16x32_bf16(
                /*neg_a=*/false, av, /*neg_b=*/false, bv,
                /*c_mod=*/(short)0, c, /*reuse_a=*/false, /*reuse_b=*/false);
    }

    // ---- epilogue: bias + sigmoid, coalesced row stores ----
    // C/D layout: VGPR r -> lanes 0-15: (M=r, N=lane); lanes 16-31: (M=r+8, N=lane-16)
    const int   n  = n0 + nl;
    const float bv = bias[n];
    #pragma unroll
    for (int r = 0; r < 8; ++r) {
        const int mm = r + hi * 8;
        float v = c[r] + bv;
        v = 1.0f / (1.0f + __expf(-v));
        out[(size_t)(node_base + mm) * OUTD + n] = v;
    }
}

extern "C" void kernel_launch(void* const* d_in, const int* in_sizes, int n_in,
                              void* d_out, int out_size, void* d_ws, size_t ws_size,
                              hipStream_t stream) {
    const float* x    = (const float*)d_in[0];   // node_feat_input [N,128]
    const int*   adj  = (const int*)  d_in[1];   // adjacency [E,2]
    // d_in[2] = indices (unused by the reference computation)
    const float* w    = (const float*)d_in[3];   // weight [256,128]
    const float* bias = (const float*)d_in[4];   // bias [128]
    float* out        = (float*)d_out;

    const int n_nodes = in_sizes[2];             // == N (100000, divisible by 16)
    const int tiles   = n_nodes / 16;

    const size_t wt_bytes = (size_t)K2 * OUTD * sizeof(unsigned short);  // 64 KB
    if (ws_size >= wt_bytes) {
        unsigned short* wtg = (unsigned short*)d_ws;
        wt_prep_kernel<<<dim3((K2 * OUTD) / 256), dim3(256), 0, stream>>>(w, wtg);
        gcn_fused_kernel<true><<<dim3(tiles), dim3(256), 0, stream>>>(
            x, adj, wtg, w, bias, out, n_nodes);
    } else {
        gcn_fused_kernel<false><<<dim3(tiles), dim3(256), 0, stream>>>(
            x, adj, nullptr, w, bias, out, n_nodes);
    }
}
